// mysnn_50062138802871
// MI455X (gfx1250) — matrix operand order repaired and checked by reading, not compiled, not run on hardware
//
#include <hip/hip_runtime.h>
#include <hip/hip_bf16.h>

typedef __attribute__((ext_vector_type(8)))  __bf16 v8bf;
typedef __attribute__((ext_vector_type(16))) __bf16 v16bf;
typedef __attribute__((ext_vector_type(8)))  float  v8f;

#define NS 25600           // B*T samples (GEMM M dimension now)
#define TT 200
#define BB 128
#define DD 784
#define HH 500
#define OO 10
#define KSTR 800           // activation row stride (bf16 elements)
#define MSTR 832           // Z row stride (f32 elements)
#define DECAY_M 0.7788007830714049f   // exp(-1/4)
#define EPSBN 1e-5f

static __device__ __forceinline__ unsigned short f2bf(float f) {
  union { float f; unsigned int u; } c; c.f = f;
  unsigned int lsb = (c.u >> 16) & 1u;
  c.u += 0x7fffu + lsb;                 // round-to-nearest-even
  return (unsigned short)(c.u >> 16);
}
static __device__ __forceinline__ float bf2f(unsigned short s) {
  union { unsigned int u; float f; } c; c.u = ((unsigned int)s) << 16;
  return c.f;
}

// ---------------- weight convert f32 -> padded bf16 [Mp x Kp] ----------------
__global__ void wconv(const float* __restrict__ src, unsigned short* __restrict__ dst,
                      int M, int K, int Mp, int Kp) {
  int idx = blockIdx.x * 256 + threadIdx.x;
  if (idx >= Mp * Kp) return;
  int m = idx / Kp, k = idx - m * Kp;
  float v = (m < M && k < K) ? src[m * K + k] : 0.0f;
  dst[idx] = f2bf(v);
}

// ------- LDS tile transpose: src[b][r<R][t<200] f32 -> dst[b][t][r<Rp] bf16 ---
// grid: (ceil(200/32), Rp/32, B), block: (32, 8)
__global__ void transpose_bf16(const float* __restrict__ src, unsigned short* __restrict__ dst,
                               int R, int Rp) {
  __shared__ float tile[32][33];
  const int b = blockIdx.z;
  const int t0 = blockIdx.x * 32, r0 = blockIdx.y * 32;
  const int tx = threadIdx.x, ty = threadIdx.y;
  #pragma unroll
  for (int i = 0; i < 4; ++i) {
    int r = r0 + ty + i * 8;
    int t = t0 + tx;
    float v = 0.0f;
    if (r < R && t < TT) v = src[((size_t)b * R + r) * TT + t];
    tile[ty + i * 8][tx] = v;
  }
  __syncthreads();
  #pragma unroll
  for (int i = 0; i < 4; ++i) {
    int t = t0 + ty + i * 8;
    int r = r0 + tx;
    if (t < TT) dst[((size_t)b * TT + t) * Rp + r] = f2bf(tile[tx][ty + i * 8]);
  }
}

// ---------------- LDS-free bf16 WMMA GEMM --------------------------------------
// Z[n][o] = sum_k act[n][k] * W[o][k];  act:[NS x KSTR], W:[Mp x ldb], Z:[NS x MSTR]
// wave tile = 32 samples x 64 channels: A frags = 2x2 b128, B frags = 4x 32B,
// 8 x v_wmma_f32_16x16x32_bf16 per K-step. Operand reuse via WGP$/192MB L2.
__global__ __launch_bounds__(256)
void gemm_wmma(const unsigned short* __restrict__ Aact,
               const unsigned short* __restrict__ Bw,
               float* __restrict__ Cz, int K, int ldb, int otiles) {
  const int wid  = threadIdx.x >> 5;
  const int lane = threadIdx.x & 31;
  const int w    = blockIdx.x * 8 + wid;
  const int st   = w / otiles;           // 32-sample tile
  const int ot   = w - st * otiles;      // 64-channel tile
  const int l15 = lane & 15, h = lane >> 4;
  const int s0 = st * 32, o0 = ot * 64;

  const unsigned short* a0 = Aact + (size_t)(s0 + l15) * KSTR;
  const unsigned short* a1 = Aact + (size_t)(s0 + 16 + l15) * KSTR;
  const unsigned short* bp = Bw + (size_t)(o0 + l15) * ldb;

  v8f acc[2][4];
  #pragma unroll
  for (int u = 0; u < 2; ++u)
    #pragma unroll
    for (int s = 0; s < 4; ++s) acc[u][s] = v8f{};

  for (int k0 = 0; k0 < K; k0 += 32) {
    // A fragment: lane(m=l15,h): 16 consecutive k at k0+16h
    v8bf al0 = *(const v8bf*)(a0 + k0 + 16 * h);
    v8bf ah0 = *(const v8bf*)(a0 + k0 + 16 * h + 8);
    v8bf al1 = *(const v8bf*)(a1 + k0 + 16 * h);
    v8bf ah1 = *(const v8bf*)(a1 + k0 + 16 * h + 8);
    v16bf af0 = __builtin_shufflevector(al0, ah0, 0,1,2,3,4,5,6,7,8,9,10,11,12,13,14,15);
    v16bf af1 = __builtin_shufflevector(al1, ah1, 0,1,2,3,4,5,6,7,8,9,10,11,12,13,14,15);
    #pragma unroll
    for (int s = 0; s < 4; ++s) {
      // B fragment: lane(n=l15,h): 16 consecutive k at k0+16h, row o = o0+s*16+l15
      v16bf bf = *(const v16bf*)(bp + (size_t)s * 16 * ldb + k0 + 16 * h);
      acc[0][s] = __builtin_amdgcn_wmma_f32_16x16x32_bf16(
          false, af0, false, bf, (short)0, acc[0][s], false, false);
      acc[1][s] = __builtin_amdgcn_wmma_f32_16x16x32_bf16(
          false, af1, false, bf, (short)0, acc[1][s], false, false);
    }
  }
  // D layout: vgpr r -> sample = r + 8h, lane&15 -> channel
  #pragma unroll
  for (int u = 0; u < 2; ++u) {
    #pragma unroll
    for (int s = 0; s < 4; ++s) {
      float* crow = Cz + (size_t)(s0 + u * 16 + 8 * h) * MSTR + o0 + s * 16 + l15;
      #pragma unroll
      for (int r = 0; r < 8; ++r) crow[(size_t)r * MSTR] = acc[u][s][r];
    }
  }
}

// -------- deterministic two-stage BN stats over Z[NS x MSTR], channel-coalesced
#define CHUNK 400
#define NCHUNK 64
__global__ void bnstats_part(const float* __restrict__ Z, float* __restrict__ ps,
                             float* __restrict__ pss) {
  int o = blockIdx.x * 256 + threadIdx.x;
  int c = blockIdx.y;
  if (o >= MSTR) return;
  const float* p = Z + (size_t)c * CHUNK * MSTR + o;
  float s = 0.f, ss = 0.f;
  for (int n = 0; n < CHUNK; ++n) { float v = p[(size_t)n * MSTR]; s += v; ss += v * v; }
  ps[c * MSTR + o] = s; pss[c * MSTR + o] = ss;
}
__global__ void bnstats_final(const float* __restrict__ ps, const float* __restrict__ pss,
                              float* __restrict__ mean, float* __restrict__ istd) {
  int o = blockIdx.x * 256 + threadIdx.x;
  if (o >= DD) return;
  float s = 0.f, ss = 0.f;
  for (int c = 0; c < NCHUNK; ++c) { s += ps[c * MSTR + o]; ss += pss[c * MSTR + o]; }
  float m = s / (float)NS;
  float var = ss / (float)NS - m * m;
  mean[o] = m; istd[o] = rsqrtf(var + EPSBN);
}

// ---- BN + sigmoid + axon1 scan; lanes = consecutive channels (coalesced per t)
__global__ void bn_sig_axon(const float* __restrict__ Z, const float* __restrict__ mean,
                            const float* __restrict__ istd, const float* __restrict__ gamma,
                            const float* __restrict__ beta, const float* __restrict__ a1,
                            const float* __restrict__ a2, unsigned short* __restrict__ out) {
  int idx = blockIdx.x * 256 + threadIdx.x;   // BB * KSTR threads
  int b = idx / KSTR, o = idx - b * KSTR;
  const float* zcol = Z + (size_t)b * TT * MSTR + o;
  unsigned short* ocol = out + (size_t)b * TT * KSTR + o;
  if (o >= DD) {                               // writes the K-pad columns = 0
    for (int t = 0; t < TT; ++t) ocol[(size_t)t * KSTR] = 0;
    return;
  }
  const float mu = mean[o], is = istd[o], g = gamma[o], be = beta[o];
  const float c1 = a1[o], c2 = a2[o];
  float p1 = 0.f, p2 = 0.f;
  for (int t = 0; t < TT; ++t) {
    float x = g * (zcol[(size_t)t * MSTR] - mu) * is + be;
    float sg = 1.0f / (1.0f + expf(-x));
    float psp = c1 * p1 + c2 * p2 + sg;
    p2 = p1; p1 = psp;
    ocol[(size_t)t * KSTR] = f2bf(psp);
  }
}

// ---- LIF (hard reset) + dropout mask + next axon; maskT is [B][T][512] bf16
__global__ void lif_mask_axon(const float* __restrict__ Z, const float* __restrict__ bias,
                              const float* __restrict__ a1, const float* __restrict__ a2,
                              const unsigned short* __restrict__ maskT,
                              unsigned short* __restrict__ out) {
  int idx = blockIdx.x * 256 + threadIdx.x;   // BB * 512 threads
  int b = idx >> 9, o = idx & 511;
  const float* zcol = Z + (size_t)b * TT * MSTR + o;
  unsigned short* ocol = out + (size_t)b * TT * KSTR + o;
  if (o >= HH) {                               // keep K-pad columns zero
    for (int t = 0; t < TT; ++t) ocol[(size_t)t * KSTR] = 0;
    return;
  }
  const unsigned short* mcol = maskT + (size_t)b * TT * 512 + o;
  const float bi = bias[o], c1 = a1[o], c2 = a2[o];
  float v = 0.f, s = 0.f, p1 = 0.f, p2 = 0.f;
  for (int t = 0; t < TT; ++t) {
    v = DECAY_M * v * (1.0f - s) + zcol[(size_t)t * MSTR] + bi;
    s = (v > 1.0f) ? 1.0f : 0.0f;
    float sm = s * bf2f(mcol[(size_t)t * 512]);
    float psp = c1 * p1 + c2 * p2 + sm;
    p2 = p1; p1 = psp;
    ocol[(size_t)t * KSTR] = f2bf(psp);
  }
}

// ---------------- final LIF -> d_out [B,10,T] --------------------------------
__global__ void lif_out(const float* __restrict__ Z, const float* __restrict__ bias,
                        float* __restrict__ out) {
  int idx = blockIdx.x * 256 + threadIdx.x;   // BB * 16 threads
  if (idx >= BB * 16) return;
  int b = idx >> 4, o = idx & 15;
  if (o >= OO) return;
  const float* zcol = Z + (size_t)b * TT * MSTR + o;
  float* orow = out + ((size_t)b * OO + o) * TT;
  const float bi = bias[o];
  float v = 0.f, s = 0.f;
  for (int t = 0; t < TT; ++t) {
    v = DECAY_M * v * (1.0f - s) + zcol[(size_t)t * MSTR] + bi;
    s = (v > 1.0f) ? 1.0f : 0.0f;
    orow[t] = s;
  }
}

extern "C" void kernel_launch(void* const* d_in, const int* in_sizes, int n_in,
                              void* d_out, int out_size, void* d_ws, size_t ws_size,
                              hipStream_t stream) {
  const float* inputs = (const float*)d_in[0];
  const float* W_mlp  = (const float*)d_in[1];
  // d_in[2] = b_mlp: cancels exactly under training-mode batch-norm
  const float* gamma  = (const float*)d_in[3];
  const float* beta   = (const float*)d_in[4];
  const float* a1_1   = (const float*)d_in[5];
  const float* a2_1   = (const float*)d_in[6];
  const float* W1     = (const float*)d_in[7];
  const float* b1     = (const float*)d_in[8];
  const float* a1_2   = (const float*)d_in[9];
  const float* a2_2   = (const float*)d_in[10];
  const float* W2     = (const float*)d_in[11];
  const float* b2     = (const float*)d_in[12];
  const float* a1_3   = (const float*)d_in[13];
  const float* a2_3   = (const float*)d_in[14];
  const float* W3     = (const float*)d_in[15];
  const float* b3     = (const float*)d_in[16];
  const float* mask1  = (const float*)d_in[17];
  const float* mask2  = (const float*)d_in[18];
  float* out = (float*)d_out;

  // workspace layout (all offsets multiples of 256 bytes), total ~173 MiB
  char* ws = (char*)d_ws;
  unsigned short* wmlp = (unsigned short*)(ws);                 // 832x800 bf16
  unsigned short* w1   = (unsigned short*)(ws + 1331200);       // 512x800 bf16
  unsigned short* w2   = (unsigned short*)(ws + 2150400);       // 512x512 bf16
  unsigned short* w3   = (unsigned short*)(ws + 2674688);       // 64x512  bf16
  unsigned short* act  = (unsigned short*)(ws + 2740224);       // NS x 800 bf16
  float*          Zf   = (float*)(ws + 43700224);               // NS x 832 f32
  unsigned short* mT1  = (unsigned short*)(ws + 128897024);     // B x T x 512 bf16
  unsigned short* mT2  = (unsigned short*)(ws + 155111424);     // B x T x 512 bf16
  float*          ps   = (float*)(ws + 181325824);              // 64 x 832 f32
  float*          pss  = (float*)(ws + 181538816);              // 64 x 832 f32
  float*          mean = (float*)(ws + 181751808);              // 832 f32
  float*          istd = (float*)(ws + 181755136);              // 832 f32

  // 1. weights -> padded bf16 (row-major = K-contiguous, feeds B fragments raw)
  wconv<<<(832 * 800 + 255) / 256, 256, 0, stream>>>(W_mlp, wmlp, 784, 784, 832, 800);
  wconv<<<(512 * 800 + 255) / 256, 256, 0, stream>>>(W1,    w1,   500, 784, 512, 800);
  wconv<<<(512 * 512 + 255) / 256, 256, 0, stream>>>(W2,    w2,   500, 500, 512, 512);
  wconv<<<( 64 * 512 + 255) / 256, 256, 0, stream>>>(W3,    w3,    10, 500,  64, 512);

  // 2. restage inputs sample-major (incl. zero K-pad) and pre-transpose masks
  transpose_bf16<<<dim3(7, 25, BB), dim3(32, 8), 0, stream>>>(inputs, act, DD, KSTR);
  transpose_bf16<<<dim3(7, 16, BB), dim3(32, 8), 0, stream>>>(mask1, mT1, HH, 512);
  transpose_bf16<<<dim3(7, 16, BB), dim3(32, 8), 0, stream>>>(mask2, mT2, HH, 512);

  // 3. layer 0: x = inputs x W_mlp^T   (waves = (NS/32)*13 = 10400 -> 1300 blocks)
  gemm_wmma<<<1300, 256, 0, stream>>>(act, wmlp, Zf, 800, 800, 13);
  bnstats_part<<<dim3(4, NCHUNK), 256, 0, stream>>>(Zf, ps, pss);
  bnstats_final<<<4, 256, 0, stream>>>(ps, pss, mean, istd);
  bn_sig_axon<<<(BB * KSTR) / 256, 256, 0, stream>>>(Zf, mean, istd, gamma, beta, a1_1, a2_1, act);

  // 4. layer 1: LIF(W1 h) * mask1 -> axon2   (waves = 800*8 -> 800 blocks)
  gemm_wmma<<<800, 256, 0, stream>>>(act, w1, Zf, 800, 800, 8);
  lif_mask_axon<<<(BB * 512) / 256, 256, 0, stream>>>(Zf, b1, a1_2, a2_2, mT1, act);

  // 5. layer 2
  gemm_wmma<<<800, 256, 0, stream>>>(act, w2, Zf, 512, 512, 8);
  lif_mask_axon<<<(BB * 512) / 256, 256, 0, stream>>>(Zf, b2, a1_3, a2_3, mT2, act);

  // 6. output layer: (waves = 800*1 -> 100 blocks)
  gemm_wmma<<<100, 256, 0, stream>>>(act, w3, Zf, 512, 512, 1);
  lif_out<<<(BB * 16 + 255) / 256, 256, 0, stream>>>(Zf, b3, out);
}
